// MoEGate_52003464020209
// MI455X (gfx1250) — matrix operand order repaired and checked by reading, not compiled, not run on hardware
//
#include <hip/hip_runtime.h>
#include <hip/hip_bf16.h>
#include <stdint.h>

typedef __attribute__((ext_vector_type(16))) __bf16 v16bf;
typedef __attribute__((ext_vector_type(8)))  float  v8f;

#define H_DIM   2048
#define E_DIM   64
#define KCHUNKS 64          // 2048 / 32
#define WPACK_DW 65536      // 64 chunks * 4 etiles * 32 lanes * 8 dwords

__device__ __forceinline__ uint32_t rne_bf16_bits(float f) {
    uint32_t u = __float_as_uint(f);
    u += 0x7FFFu + ((u >> 16) & 1u);
    return u >> 16;
}

// Split float pair (even K, odd K) into packed bf16 hi dword and residual-lo dword.
__device__ __forceinline__ void split_pair(float f0, float f1, uint32_t& hw, uint32_t& lw) {
    uint32_t b0 = __float_as_uint(f0), b1 = __float_as_uint(f1);
    uint32_t h0 = b0 & 0xFFFF0000u,   h1 = b1 & 0xFFFF0000u;   // truncated bf16 (exact)
    hw = (h0 >> 16) | h1;
    float l0 = f0 - __uint_as_float(h0);                       // exact residual
    float l1 = f1 - __uint_as_float(h1);
    lw = rne_bf16_bits(l0) | (rne_bf16_bits(l1) << 16);
}

// ---------------------------------------------------------------------------
// Pre-kernel: split W[64][2048] fp32 into bf16 hi/lo, pre-swizzled into the
// WMMA B-operand layout: out[((c*4+et)*32+lane)*8 + r] holds packed pair
//   K = c*32 + (lane>>4)*8 + 2r + (r>=4 ? 8 : 0) (+1),  N(expert) = et*16 + (lane&15)
// so every lane's 8 B dwords are contiguous (two b128 loads in main kernel).
// ---------------------------------------------------------------------------
__global__ void moe_gate_split_w(const float* __restrict__ w,
                                 uint32_t* __restrict__ whi,
                                 uint32_t* __restrict__ wlo) {
    int tid  = blockIdx.x * blockDim.x + threadIdx.x;   // 0 .. 65535
    int r    = tid & 7;
    int lane = (tid >> 3) & 31;
    int et   = (tid >> 8) & 3;
    int c    = tid >> 10;
    int k = c * 32 + (lane >> 4) * 8 + 2 * r + (r >= 4 ? 8 : 0);
    int e = et * 16 + (lane & 15);
    float w0 = w[e * H_DIM + k];
    float w1 = w[e * H_DIM + k + 1];
    uint32_t hw, lw;
    split_pair(w0, w1, hw, lw);
    whi[tid] = hw;
    wlo[tid] = lw;
}

// ---------------------------------------------------------------------------
// Main kernel: 8 waves/block, each wave computes a 16-token x 64-expert logit
// tile with bf16x3 WMMA (Ah*Bh + Ah*Bl + Al*Bh), then top-6 + softmax.
// ---------------------------------------------------------------------------
__global__ void __launch_bounds__(256)
moe_gate_kernel(const float* __restrict__ x,
                const uint32_t* __restrict__ whi,
                const uint32_t* __restrict__ wlo,
                int* __restrict__ idx_out,
                float* __restrict__ wt_out,
                int T) {
    __shared__ float slog[8 * 16 * 68];   // per-wave 16 rows x 64 (+4 pad) logits

    const int tid  = threadIdx.x;
    const int wave = tid >> 5;
    const int lane = tid & 31;
    const int m    = lane & 15;           // token row within tile / expert col
    const int half = lane >> 4;           // K-half selector
    const int row0 = blockIdx.x * 128 + wave * 16;
    if (row0 + 16 > T) return;            // whole-wave guard (EXEC stays all-1s)

    const float* rowp = x + (size_t)(row0 + m) * H_DIM;

    v8f acc0 = 0.f, acc1 = 0.f, acc2 = 0.f, acc3 = 0.f;

    for (int c = 0; c < KCHUNKS; ++c) {
        const int k0 = c * 32;
        // stream-ahead prefetch of x (speculative, non-faulting)
        __builtin_prefetch(rowp + k0 + 256, 0, 0);

        // ---- A operand: lanes 0-15 regs0-3 K=0..7, regs4-7 K=16..23;
        //                 lanes 16-31 shifted by +8 (ISA 16-bit A layout)
        float4 a0 = *(const float4*)(rowp + k0 + half * 8);
        float4 a1 = *(const float4*)(rowp + k0 + half * 8 + 4);
        float4 a2 = *(const float4*)(rowp + k0 + 16 + half * 8);
        float4 a3 = *(const float4*)(rowp + k0 + 16 + half * 8 + 4);

        union { uint32_t u[8]; v16bf v; } Ah, Al;
        split_pair(a0.x, a0.y, Ah.u[0], Al.u[0]);
        split_pair(a0.z, a0.w, Ah.u[1], Al.u[1]);
        split_pair(a1.x, a1.y, Ah.u[2], Al.u[2]);
        split_pair(a1.z, a1.w, Ah.u[3], Al.u[3]);
        split_pair(a2.x, a2.y, Ah.u[4], Al.u[4]);
        split_pair(a2.z, a2.w, Ah.u[5], Al.u[5]);
        split_pair(a3.x, a3.y, Ah.u[6], Al.u[6]);
        split_pair(a3.z, a3.w, Ah.u[7], Al.u[7]);

        const uint32_t bidx = (uint32_t)((c * 4) * 32 + lane) * 8u;

#define ETILE(ET, ACC)                                                          \
        {                                                                       \
            union { uint4 q[2]; v16bf v; } Bh, Bl;                              \
            const uint32_t off = bidx + (ET) * 256u;                            \
            Bh.q[0] = *(const uint4*)(whi + off);                               \
            Bh.q[1] = *(const uint4*)(whi + off + 4);                           \
            Bl.q[0] = *(const uint4*)(wlo + off);                               \
            Bl.q[1] = *(const uint4*)(wlo + off + 4);                           \
            ACC = __builtin_amdgcn_wmma_f32_16x16x32_bf16(                      \
                false, Ah.v, false, Bh.v, (short)0, ACC, false, false);         \
            ACC = __builtin_amdgcn_wmma_f32_16x16x32_bf16(                      \
                false, Ah.v, false, Bl.v, (short)0, ACC, false, false);         \
            ACC = __builtin_amdgcn_wmma_f32_16x16x32_bf16(                      \
                false, Al.v, false, Bh.v, (short)0, ACC, false, false);         \
        }
        ETILE(0, acc0)
        ETILE(1, acc1)
        ETILE(2, acc2)
        ETILE(3, acc3)
#undef ETILE
    }

    // ---- spill logits to LDS. C layout: lanes 0-15 -> M=v, lanes 16-31 -> M=v+8, N=lane&15
    const int base = wave * 16 * 68;
#pragma unroll
    for (int v = 0; v < 8; ++v) {
        const int M = v + half * 8;
        slog[base + M * 68 +  0 + m] = acc0[v];
        slog[base + M * 68 + 16 + m] = acc1[v];
        slog[base + M * 68 + 32 + m] = acc2[v];
        slog[base + M * 68 + 48 + m] = acc3[v];
    }
    __syncthreads();

    // ---- top-6 + softmax: lanes 0-15 each own one token row
    if (lane < 16) {
        const float* row = &slog[base + lane * 68];
        float bv[6];
        int   bi[6];
#pragma unroll
        for (int j = 0; j < 6; ++j) { bv[j] = -__builtin_inff(); bi[j] = 0; }
        for (int e = 0; e < E_DIM; ++e) {
            float v = row[e];
            if (v > bv[5]) {                      // strict >: equal keeps lower index
                int j = 5;
                while (j > 0 && v > bv[j - 1]) {
                    bv[j] = bv[j - 1]; bi[j] = bi[j - 1]; --j;
                }
                bv[j] = v; bi[j] = e;
            }
        }
        float p[6], s = 0.f;
#pragma unroll
        for (int j = 0; j < 6; ++j) { p[j] = __expf(bv[j] - bv[0]); s += p[j]; }
        float inv = 1.f / s;
        float s2 = 0.f;
#pragma unroll
        for (int j = 0; j < 6; ++j) { p[j] *= inv; s2 += p[j]; }
        float inv2 = 1.f / (s2 + 1e-20f);         // mirror reference renorm
        const size_t t = (size_t)(row0 + lane);
#pragma unroll
        for (int j = 0; j < 6; ++j) {
            idx_out[t * 6 + j] = bi[j];
            wt_out [t * 6 + j] = p[j] * inv2;
        }
    }
}

extern "C" void kernel_launch(void* const* d_in, const int* in_sizes, int n_in,
                              void* d_out, int out_size, void* d_ws, size_t ws_size,
                              hipStream_t stream) {
    const float* x = (const float*)d_in[0];   // [4,4096,2048] fp32
    const float* w = (const float*)d_in[1];   // [64,2048] fp32
    const int T = in_sizes[0] / H_DIM;        // 16384 tokens

    uint32_t* whi = (uint32_t*)d_ws;          // 256 KB
    uint32_t* wlo = whi + WPACK_DW;           // 256 KB

    int*   idx_out = (int*)d_out;                       // [T,6] int32
    float* wt_out  = (float*)d_out + (size_t)T * 6;     // [T,6] fp32

    moe_gate_split_w<<<WPACK_DW / 256, 256, 0, stream>>>(w, whi, wlo);
    moe_gate_kernel<<<T / 128, 256, 0, stream>>>(x, whi, wlo, idx_out, wt_out, T);
}
